// SwitchMoE_1967095021955
// MI455X (gfx1250) — matrix operand order repaired and checked by reading, not compiled, not run on hardware
//
#include <hip/hip_runtime.h>
#include <math.h>

// ---------------- problem dims ----------------
#define BDIM 2
#define TDIM 2048
#define NTOK (BDIM * TDIM)   // 4096 tokens
#define EEXP 8
#define CDIM 1024
#define HDIM 2048
#define GDIM 16

typedef __bf16 bf16_t;
typedef __attribute__((ext_vector_type(16))) __bf16 v16bf;
typedef __attribute__((ext_vector_type(8)))  float  v8f;

union Frag16 {          // one WMMA 16-bit operand: 16 bf16 per lane = 32 bytes
    v16bf v;
    uint4 q[2];
};

static __device__ __forceinline__ v8f wmma_bf16(const Frag16& a, const Frag16& b, v8f c) {
    return __builtin_amdgcn_wmma_f32_16x16x32_bf16(
        /*neg_a=*/false, a.v, /*neg_b=*/false, b.v,
        /*c_mod=*/(short)0, c, /*reuse_a=*/false, /*reuse_b=*/false);
}

// ---------------- init: zero counters, fill token lists with -1 ----------------
__global__ void k_init(int* __restrict__ counts, int* __restrict__ list, int n) {
    int i = blockIdx.x * blockDim.x + threadIdx.x;
    if (i < EEXP) counts[i] = 0;
    if (i < n)    list[i] = -1;
}

// ---------------- f32 -> bf16 row copy (x) ----------------
__global__ void k_cvt_x(const float* __restrict__ in, bf16_t* __restrict__ out, int n) {
    int i = (blockIdx.x * blockDim.x + threadIdx.x) * 4;
    if (i >= n) return;
    float4 v = *(const float4*)(in + i);
    union { bf16_t b[4]; uint2 u; } p;
    p.b[0] = (bf16_t)v.x; p.b[1] = (bf16_t)v.y;
    p.b[2] = (bf16_t)v.z; p.b[3] = (bf16_t)v.w;
    *(uint2*)(out + i) = p.u;
}

// ---------------- w_red [C][G] f32 -> wredT [G][C] bf16 (tiny) ----------------
__global__ void k_cvt_wred(const float* __restrict__ in, bf16_t* __restrict__ out) {
    int i = blockIdx.x * blockDim.x + threadIdx.x;
    if (i >= CDIM * GDIM) return;
    int c = i / GDIM, g = i % GDIM;
    out[g * CDIM + c] = (bf16_t)in[i];
}

// ---------------- per-expert transpose+convert: in[e][R][Cc] f32 -> out[e][Cc][R] bf16 ----
__global__ void k_transpose_cvt(const float* __restrict__ in, bf16_t* __restrict__ out,
                                int R, int Cc) {
    __shared__ float tile[32][33];
    const size_t esz = (size_t)R * Cc;
    in  += (size_t)blockIdx.z * esz;
    out += (size_t)blockIdx.z * esz;
    int c0 = blockIdx.x * 32, r0 = blockIdx.y * 32;
#pragma unroll
    for (int i = threadIdx.y; i < 32; i += 8) {
        tile[i][threadIdx.x] = in[(size_t)(r0 + i) * Cc + (c0 + threadIdx.x)];
    }
    __syncthreads();
#pragma unroll
    for (int i = threadIdx.y; i < 32; i += 8) {
        out[(size_t)(c0 + i) * R + (r0 + threadIdx.x)] = (bf16_t)tile[threadIdx.x][i];
    }
}

// ---------------- gate: WMMA x_red, logits vs normalized wg, top-1 routing ---------
__global__ __launch_bounds__(32) void k_gate(const bf16_t* __restrict__ xbf,
                                             const bf16_t* __restrict__ wredT,
                                             const float* __restrict__ wg,
                                             float* __restrict__ score,
                                             int* __restrict__ counts,
                                             int* __restrict__ list) {
    __shared__ float xr[16][17];     // x_red tile [M=token][N=g]
    __shared__ float wgn[EEXP][GDIM];
    int lane = threadIdx.x;
    if (lane < EEXP) {
        float s = 0.f;
#pragma unroll
        for (int g = 0; g < GDIM; ++g) { float v = wg[lane * GDIM + g]; s += v * v; }
        float inv = 1.f / fmaxf(sqrtf(s), 1e-4f);
#pragma unroll
        for (int g = 0; g < GDIM; ++g) wgn[lane][g] = wg[lane * GDIM + g] * inv;
    }
    int tile = blockIdx.x;
    int m = lane & 15, h = lane >> 4;
    const bf16_t* arow = xbf + (size_t)(tile * 16 + m) * CDIM;   // A: lane = M (token)
    const bf16_t* brow = wredT + (size_t)m * CDIM;               // B: lane = N (= g)
    v8f acc;
#pragma unroll
    for (int r = 0; r < 8; ++r) acc[r] = 0.f;
    for (int k = 0; k < CDIM; k += 32) {
        Frag16 a, b;
        a.q[0] = *(const uint4*)(arow + k + h * 8);
        a.q[1] = *(const uint4*)(arow + k + 16 + h * 8);
        b.q[0] = *(const uint4*)(brow + k + h * 8);
        b.q[1] = *(const uint4*)(brow + k + 16 + h * 8);
        acc = wmma_bf16(a, b, acc);
    }
#pragma unroll
    for (int r = 0; r < 8; ++r) xr[r + 8 * h][m] = acc[r];       // D: M=r+8h, N=lane&15
    __syncthreads();
    if (lane < 16) {
        int tok = tile * 16 + lane;
        float logits[EEXP];
        float mx = -1e30f;
#pragma unroll
        for (int e = 0; e < EEXP; ++e) {
            float s = 0.f;
#pragma unroll
            for (int g = 0; g < GDIM; ++g) s += xr[lane][g] * wgn[e][g];
            logits[e] = s;
            mx = fmaxf(mx, s);
        }
        float den = 0.f, bl = -1e30f; int best = 0;
#pragma unroll
        for (int e = 0; e < EEXP; ++e) {
            den += expf(logits[e] - mx);
            if (logits[e] > bl) { bl = logits[e]; best = e; }
        }
        score[tok] = expf(bl - mx) / den;      // top-1 softmax gate
        int pos = atomicAdd(&counts[best], 1);
        list[best * NTOK + pos] = tok;
    }
}

// ---------------- fused expert FFN: GEMM1 + bias + erf-GELU + GEMM2 + bias + scale ---
__global__ __launch_bounds__(256) void k_moe(const bf16_t* __restrict__ xbf,
                                             const bf16_t* __restrict__ w1t,  // [E][H][C]
                                             const bf16_t* __restrict__ w2t,  // [E][C][H]
                                             const float* __restrict__ b1,
                                             const float* __restrict__ b2,
                                             const float* __restrict__ score,
                                             const int* __restrict__ counts,
                                             const int* __restrict__ list,
                                             float* __restrict__ out) {
    const int e = blockIdx.y;
    const int cnt = counts[e];
    const int ntiles = (cnt + 15) >> 4;
    if ((int)blockIdx.x >= ntiles) return;    // uniform per-block exit

    __shared__ __align__(16) bf16_t hbuf[16][CDIM / 1 + 8 > 0 ? 1032 : 1032]; // 16 x 1032 bf16
    __shared__ int   toks[16];
    __shared__ float scs[16];

    const int tid = threadIdx.x;
    const int lane = tid & 31;
    const int w = tid >> 5;                   // wave id 0..7
    if (tid < 16) {
        int t = list[e * NTOK + blockIdx.x * 16 + tid];
        toks[tid] = t;
        scs[tid] = (t >= 0) ? score[t] : 0.f;
    }
    __syncthreads();

    const int m = lane & 15, h = lane >> 4;
    const int tA = toks[m];
    const bf16_t* arow = xbf + (size_t)(tA < 0 ? 0 : tA) * CDIM;  // A rows for GEMM1

    // persistent GEMM2 accumulators: 8 column tiles per wave (64 tiles of 16 over C)
    v8f acc2[8];
#pragma unroll
    for (int j = 0; j < 8; ++j) {
        float bb = b2[e * CDIM + (w + 8 * j) * 16 + m];
#pragma unroll
        for (int r = 0; r < 8; ++r) acc2[j][r] = bb;
    }

    for (int half = 0; half < 2; ++half) {
        const int hbase = half * (HDIM / 2);
        // ---- GEMM1 over this H-half: 64 n-tiles / 8 waves = 8 per wave ----
        for (int j = 0; j < 8; ++j) {
            const int n0 = hbase + (w + 8 * j) * 16;
            v8f acc;
            float bb = b1[e * HDIM + n0 + m];
#pragma unroll
            for (int r = 0; r < 8; ++r) acc[r] = bb;
            const bf16_t* brow = w1t + ((size_t)e * HDIM + n0 + m) * CDIM;
            for (int k = 0; k < CDIM; k += 32) {
                Frag16 a, b;
                a.q[0] = *(const uint4*)(arow + k + h * 8);
                a.q[1] = *(const uint4*)(arow + k + 16 + h * 8);
                b.q[0] = *(const uint4*)(brow + k + h * 8);
                b.q[1] = *(const uint4*)(brow + k + 16 + h * 8);
                acc = wmma_bf16(a, b, acc);
            }
            const int nl = (n0 - hbase) + m;  // local column in hbuf
#pragma unroll
            for (int r = 0; r < 8; ++r) {
                float v = acc[r];
                float g = 0.5f * v * (1.f + erff(v * 0.70710678118654752f)); // exact GELU
                hbuf[r + 8 * h][nl] = (bf16_t)g;
            }
        }
        __syncthreads();
        // ---- GEMM2 partial accumulate over this half's K=1024 ----
        for (int j = 0; j < 8; ++j) {
            const int c0 = (w + 8 * j) * 16;
            const bf16_t* brow2 = w2t + ((size_t)e * CDIM + c0 + m) * HDIM + hbase;
            v8f acc = acc2[j];
            for (int k = 0; k < HDIM / 2; k += 32) {
                Frag16 a, b;
                a.q[0] = *(const uint4*)&hbuf[m][k + h * 8];
                a.q[1] = *(const uint4*)&hbuf[m][k + 16 + h * 8];
                b.q[0] = *(const uint4*)(brow2 + k + h * 8);
                b.q[1] = *(const uint4*)(brow2 + k + 16 + h * 8);
                acc = wmma_bf16(a, b, acc);
            }
            acc2[j] = acc;
        }
        __syncthreads();
    }

    // ---- epilogue: scale by gate score, scatter to token rows ----
#pragma unroll
    for (int j = 0; j < 8; ++j) {
        const int c0 = (w + 8 * j) * 16;
#pragma unroll
        for (int r = 0; r < 8; ++r) {
            const int M = r + 8 * h;
            const int t = toks[M];
            if (t >= 0) out[(size_t)t * CDIM + c0 + m] = acc2[j][r] * scs[M];
        }
    }
}

// ---------------- host launcher ----------------
extern "C" void kernel_launch(void* const* d_in, const int* in_sizes, int n_in,
                              void* d_out, int out_size, void* d_ws, size_t ws_size,
                              hipStream_t stream) {
    const float* x     = (const float*)d_in[0];   // [B,T,C]
    const float* w_red = (const float*)d_in[1];   // [C,G]
    const float* wg    = (const float*)d_in[2];   // [E,G]
    const float* W1    = (const float*)d_in[3];   // [E,C,H]
    const float* b1    = (const float*)d_in[4];   // [E,H]
    const float* W2    = (const float*)d_in[5];   // [E,H,C]
    const float* b2    = (const float*)d_in[6];   // [E,C]
    float* out = (float*)d_out;

    // workspace carve (~75.7 MB total)
    char* ws = (char*)d_ws;
    size_t off = 0;
    auto carve = [&](size_t bytes) -> void* {
        off = (off + 255) & ~(size_t)255;
        void* p = ws + off;
        off += bytes;
        return p;
    };
    int*    counts = (int*)   carve(EEXP * sizeof(int));
    int*    list   = (int*)   carve((size_t)EEXP * NTOK * sizeof(int));
    float*  score  = (float*) carve((size_t)NTOK * sizeof(float));
    bf16_t* xbf    = (bf16_t*)carve((size_t)NTOK * CDIM * sizeof(bf16_t));
    bf16_t* wredT  = (bf16_t*)carve((size_t)GDIM * CDIM * sizeof(bf16_t));
    bf16_t* w1t    = (bf16_t*)carve((size_t)EEXP * HDIM * CDIM * sizeof(bf16_t));
    bf16_t* w2t    = (bf16_t*)carve((size_t)EEXP * HDIM * CDIM * sizeof(bf16_t));
    (void)ws_size; (void)in_sizes; (void)n_in; (void)out_size;

    k_init<<<(EEXP * NTOK + 255) / 256, 256, 0, stream>>>(counts, list, EEXP * NTOK);
    k_cvt_x<<<(NTOK * CDIM / 4 + 255) / 256, 256, 0, stream>>>(x, xbf, NTOK * CDIM);
    k_cvt_wred<<<(CDIM * GDIM + 255) / 256, 256, 0, stream>>>(w_red, wredT);
    dim3 tb(32, 8);
    k_transpose_cvt<<<dim3(HDIM / 32, CDIM / 32, EEXP), tb, 0, stream>>>(W1, w1t, CDIM, HDIM);
    k_transpose_cvt<<<dim3(CDIM / 32, HDIM / 32, EEXP), tb, 0, stream>>>(W2, w2t, HDIM, CDIM);
    k_gate<<<NTOK / 16, 32, 0, stream>>>(xbf, wredT, wg, score, counts, list);
    k_moe<<<dim3(NTOK / 16, EEXP), 256, 0, stream>>>(xbf, w1t, w2t, b1, b2, score,
                                                     counts, list, out);
}
